// Attention_17437567221951
// MI455X (gfx1250) — matrix operand order repaired and checked by reading, not compiled, not run on hardware
//
#include <hip/hip_runtime.h>
#include <hip/hip_bf16.h>
#include <math.h>

// ---------------------------------------------------------------------------
// Problem constants (from reference setup_inputs)
// ---------------------------------------------------------------------------
#define B_     128
#define T_     512
#define DW     300
#define DH     512
#define DA     512
#define NG     2048          // 4*DH gate width
#define KX     320           // words K padded to multiple of 32
#define KTOT   832           // KX + DH   (LSTM gate GEMM K)
#define KATT   832           // DH + KX   (attention GEMM K)
#define KSTEPS 26            // 832 / 32
#define NBLK   32            // persistent LSTM grid (DH/16)
#define WLDS_BYTES (4 * KSTEPS * 16 * 32 * 2)   // 106496 B weight slice in LDS

typedef __attribute__((ext_vector_type(16))) __bf16       v16bf;
typedef __attribute__((ext_vector_type(8)))  __bf16       v8bf;
typedef __attribute__((ext_vector_type(8)))  float        v8f;
typedef __attribute__((ext_vector_type(8)))  float        v8fv;
typedef __attribute__((ext_vector_type(4)))  unsigned int u32x4;

union BFrag { v16bf v; u32x4 q[2]; };

__device__ __forceinline__ v8f wmma_bf16(const BFrag& a, const BFrag& b, v8f c) {
  return __builtin_amdgcn_wmma_f32_16x16x32_bf16(
      false, a.v, false, b.v, (short)0, c, false, false);
}

__device__ __forceinline__ float sigmoidf_(float x) {
  return 1.0f / (1.0f + __expf(-x));
}

__device__ __forceinline__ u32x4 cvt8(const float* p) {
  v8fv f = *(const v8fv*)p;
  union { v8bf v; u32x4 q; } cv;
  cv.v = __builtin_convertvector(f, v8bf);
  return cv.q;
}

// ---------------------------------------------------------------------------
// Prep kernels
// ---------------------------------------------------------------------------
__global__ void prep_words_kernel(const float* __restrict__ src,
                                  __bf16* __restrict__ dst, int n) {
  int i = blockIdx.x * 256 + threadIdx.x;
  if (i >= n) return;
  int g = i / KX, k = i - g * KX;
  dst[i] = (__bf16)((k < DW) ? src[(size_t)g * DW + k] : 0.0f);
}

__global__ void prep_wcomb_kernel(const float* __restrict__ Wih,
                                  const float* __restrict__ Whh,
                                  __bf16* __restrict__ dst, int n) {
  int i = blockIdx.x * 256 + threadIdx.x;
  if (i >= n) return;
  int r = i / KTOT, k = i - r * KTOT;
  float v;
  if (k < KX) v = (k < DW) ? Wih[(size_t)r * DW + k] : 0.0f;
  else        v = Whh[(size_t)r * DH + (k - KX)];
  dst[i] = (__bf16)v;
}

__global__ void prep_bias_kernel(const float* __restrict__ bih,
                                 const float* __restrict__ bhh,
                                 float* __restrict__ dst, int n) {
  int i = blockIdx.x * 256 + threadIdx.x;
  if (i < n) dst[i] = bih[i] + bhh[i];
}

__global__ void prep_w1t_kernel(const float* __restrict__ w1,
                                __bf16* __restrict__ dst, int n) {
  int i = blockIdx.x * 256 + threadIdx.x;
  if (i >= n) return;
  int c = i / KATT, k = i - c * KATT;
  dst[i] = (__bf16)((k < DW + DH) ? w1[(size_t)k * DA + c] : 0.0f);
}

__global__ void prep_aspect_kernel(const float* __restrict__ target,
                                   __bf16* __restrict__ dst, int n) {
  int i = blockIdx.x * 256 + threadIdx.x;
  if (i >= n) return;
  int b = i / KX, k = i - b * KX;
  dst[i] = (__bf16)((k < DW) ? target[(size_t)b * DW + k] : 0.0f);
}

__global__ void prep_state_kernel(__bf16* __restrict__ h0,
                                  __bf16* __restrict__ h1,
                                  int* __restrict__ cnt, int n) {
  int i = blockIdx.x * 256 + threadIdx.x;
  if (i >= n) return;
  h0[i] = (__bf16)0.0f;
  h1[i] = (__bf16)0.0f;
  if (i == 0) *cnt = 0;
}

// ---------------------------------------------------------------------------
// Grid barrier (monotonic counter, release/acquire at agent scope)
// ---------------------------------------------------------------------------
__device__ __forceinline__ void grid_barrier(int* cnt, int target) {
  __threadfence();
  __syncthreads();
  if (threadIdx.x == 0) {
    __hip_atomic_fetch_add(cnt, 1, __ATOMIC_RELEASE, __HIP_MEMORY_SCOPE_AGENT);
    while (__hip_atomic_load(cnt, __ATOMIC_ACQUIRE, __HIP_MEMORY_SCOPE_AGENT) <
           target) {
      __builtin_amdgcn_s_sleep(2);
    }
  }
  __syncthreads();
  __threadfence();
}

// ---------------------------------------------------------------------------
// Persistent LSTM: one launch, 512 internal steps, LDS-resident weights,
// register-resident c/h state.  grid = 32 WGs x 256 threads.
// ---------------------------------------------------------------------------
#define LOAD_A_LSTM(dst, ks_)                                                  \
  do {                                                                         \
    const int kb_ = (ks_) << 5;                                                \
    if ((ks_) < 10) {                                                          \
      dst.q[0] = *(const u32x4*)(xrow + kb_ + (half << 3));                    \
      dst.q[1] = *(const u32x4*)(xrow + kb_ + 16 + (half << 3));               \
    } else {                                                                   \
      dst.q[0] = *(const u32x4*)(hrow + (kb_ - KX) + (half << 3));             \
      dst.q[1] = *(const u32x4*)(hrow + (kb_ - KX) + 16 + (half << 3));        \
    }                                                                          \
  } while (0)

#define LOAD_B_LSTM(dst, g_, ks_)                                              \
  do {                                                                         \
    const __bf16* lb_ = wl_lane + ((((g_)*KSTEPS) + (ks_)) << 9);              \
    dst.q[0] = *(const u32x4*)(lb_);                                           \
    dst.q[1] = *(const u32x4*)(lb_ + 16);                                      \
  } while (0)

__global__ __launch_bounds__(256) void lstm_persistent_kernel(
    const __bf16* __restrict__ xw,    // [B*T][KX]
    const __bf16* __restrict__ Wc,    // [NG][KTOT]
    const float*  __restrict__ bias,  // [NG]
    const int*    __restrict__ lens,  // [B]
    __bf16*       __restrict__ hb0,   // [B][DH] ping
    __bf16*       __restrict__ hb1,   // [B][DH] pong
    float*        __restrict__ H,     // [B*T][DH]
    int*          __restrict__ cnt) {
  extern __shared__ char smem_raw[];
  __bf16* wl = (__bf16*)smem_raw;

  const int n0   = blockIdx.x << 4;
  const int wave = threadIdx.x >> 5;
  const int lane = threadIdx.x & 31;
  const int sub  = lane & 15;
  const int half = lane >> 4;

  // ---- stage this WG's 104KB weight slice into LDS (once) ----
  // layout: [(gate*KSTEPS + ks)*16 + col][32]  bf16
  for (int ci = threadIdx.x; ci < (4 * 16 * KTOT) / 8; ci += 256) {
    const int r    = ci / 104;        // gate*16 + col
    const int k    = (ci - r * 104) << 3;
    const int gate = r >> 4, col = r & 15;
    const u32x4 v =
        *(const u32x4*)(Wc + (size_t)((gate << 9) + n0 + col) * KTOT + k);
    *(u32x4*)(wl + ((((gate * KSTEPS) + (k >> 5)) * 16 + col) << 5) +
              (k & 31)) = v;
  }
  __syncthreads();

  const __bf16* wl_lane = wl + (sub << 5) + (half << 3);

  // ---- per-lane constants ----
  const int   b_a   = (wave << 4) + sub;      // A-fragment source batch row
  const int   col   = n0 + sub;               // output column (C/D layout)
  const int   rbase = (wave << 4) + (half << 3);
  const float bI_ = bias[col];
  const float bF_ = bias[512 + col];
  const float bG_ = bias[1024 + col];
  const float bO_ = bias[1536 + col];

  int len_r[8];
#pragma unroll
  for (int r = 0; r < 8; ++r) len_r[r] = lens[rbase + r];

  float cReg[8], hReg[8];
#pragma unroll
  for (int r = 0; r < 8; ++r) { cReg[r] = 0.0f; hReg[r] = 0.0f; }

  const int nwg = (int)gridDim.x;

  for (int t = 0; t < T_; ++t) {
    const __bf16* hin  = (t & 1) ? hb1 : hb0;
    __bf16*       hout = (t & 1) ? hb0 : hb1;
    const __bf16* xrow = xw + ((size_t)b_a * T_ + t) * KX;
    const __bf16* hrow = hin + (size_t)b_a * DH;

    v8f accI = {}, accF = {}, accG = {}, accO = {};

    // software-pipelined K loop: A prefetched one step ahead; B loads grouped
    BFrag aP[2];
    LOAD_A_LSTM(aP[0], 0);
#pragma unroll
    for (int ks = 0; ks < KSTEPS; ++ks) {
      const int cur = ks & 1;
      if (ks + 1 < KSTEPS) LOAD_A_LSTM(aP[cur ^ 1], ks + 1);
      BFrag bI, bF, bG, bO;
      LOAD_B_LSTM(bI, 0, ks);
      LOAD_B_LSTM(bF, 1, ks);
      LOAD_B_LSTM(bG, 2, ks);
      LOAD_B_LSTM(bO, 3, ks);
      accI = wmma_bf16(aP[cur], bI, accI);
      accF = wmma_bf16(aP[cur], bF, accF);
      accG = wmma_bf16(aP[cur], bG, accG);
      accO = wmma_bf16(aP[cur], bO, accO);
    }

    // cell update: element r is (row rbase+r, col)
#pragma unroll
    for (int r = 0; r < 8; ++r) {
      const float gi = sigmoidf_(accI[r] + bI_);
      const float gf = sigmoidf_(accF[r] + bF_);
      const float gg = tanhf(accG[r] + bG_);
      const float go = sigmoidf_(accO[r] + bO_);
      float cN = gf * cReg[r] + gi * gg;
      float hN = go * tanhf(cN);
      if (t >= len_r[r]) { cN = cReg[r]; hN = hReg[r]; }   // masked carry
      cReg[r] = cN;
      hReg[r] = hN;
      const int b = rbase + r;
      hout[(size_t)b * DH + col]         = (__bf16)hN;
      H[((size_t)b * T_ + t) * DH + col] = hN;
    }

    if (t + 1 < T_) grid_barrier(cnt, (t + 1) * nwg);
  }
}

// ---------------------------------------------------------------------------
// Attention scores: M = tanh([H|aspect] @ w1 + b1); score = M @ w (fused)
// grid = B*T/16 = 4096, block = 256 (8 waves x 4 N-blocks = 512 cols)
// ---------------------------------------------------------------------------
#define LOAD_A_ATT(dst, ks_)                                                   \
  do {                                                                         \
    const int kb_ = (ks_) << 5;                                                \
    if ((ks_) < 16) {                                                          \
      dst.q[0] = cvt8(hrow + kb_ + (half << 3));                               \
      dst.q[1] = cvt8(hrow + kb_ + 16 + (half << 3));                          \
    } else {                                                                   \
      dst.q[0] = *(const u32x4*)(arow + (kb_ - DH) + (half << 3));             \
      dst.q[1] = *(const u32x4*)(arow + (kb_ - DH) + 16 + (half << 3));        \
    }                                                                          \
  } while (0)

#define LOAD_B_ATT(dst, j_, ks_)                                               \
  do {                                                                         \
    const __bf16* wb_ = w1row + (size_t)(j_) * (16 * KATT) + ((ks_) << 5);     \
    dst.q[0] = *(const u32x4*)(wb_);                                           \
    dst.q[1] = *(const u32x4*)(wb_ + 16);                                      \
  } while (0)

__global__ __launch_bounds__(256) void attn_score_kernel(
    const float*  __restrict__ H,      // [B*T][DH]
    const __bf16* __restrict__ asp,    // [B][KX]
    const __bf16* __restrict__ w1t,    // [DA][KATT]
    const float*  __restrict__ b1,     // [DA]
    const float*  __restrict__ w,      // [DA]
    float*        __restrict__ scores) {
  __shared__ float sS[16];
  const int g0   = blockIdx.x << 4;
  const int wave = threadIdx.x >> 5;
  const int lane = threadIdx.x & 31;
  const int sub  = lane & 15;
  const int half = lane >> 4;

  if (threadIdx.x < 16) sS[threadIdx.x] = 0.0f;
  __syncthreads();

  const int     g     = g0 + sub;
  const float*  hrow  = H + (size_t)g * DH;
  const __bf16* arow  = asp + (size_t)(g >> 9) * KX;      // b = g / T
  const __bf16* w1row = w1t + (size_t)((wave << 6) + sub) * KATT + (half << 3);

  v8f acc[4] = {};
  BFrag aP[2];
  LOAD_A_ATT(aP[0], 0);
#pragma unroll
  for (int ks = 0; ks < KSTEPS; ++ks) {
    const int cur = ks & 1;
    if (ks + 1 < KSTEPS) LOAD_A_ATT(aP[cur ^ 1], ks + 1);
    BFrag b0, b1f, b2, b3;
    LOAD_B_ATT(b0, 0, ks);
    LOAD_B_ATT(b1f, 1, ks);
    LOAD_B_ATT(b2, 2, ks);
    LOAD_B_ATT(b3, 3, ks);
    acc[0] = wmma_bf16(aP[cur], b0, acc[0]);
    acc[1] = wmma_bf16(aP[cur], b1f, acc[1]);
    acc[2] = wmma_bf16(aP[cur], b2, acc[2]);
    acc[3] = wmma_bf16(aP[cur], b3, acc[3]);
  }

  // fused tanh * w + row reduce; element r -> row (8*half + r), col n0+sub
#pragma unroll
  for (int r = 0; r < 8; ++r) {
    float p = 0.0f;
#pragma unroll
    for (int j = 0; j < 4; ++j) {
      const int n = (wave << 6) + (j << 4) + sub;
      p += tanhf(acc[j][r] + b1[n]) * w[n];
    }
    p += __shfl_xor(p, 1, 32);
    p += __shfl_xor(p, 2, 32);
    p += __shfl_xor(p, 4, 32);
    p += __shfl_xor(p, 8, 32);
    if (sub == 0) atomicAdd(&sS[(half << 3) + r], p);
  }
  __syncthreads();
  if (threadIdx.x < 16) scores[g0 + threadIdx.x] = sS[threadIdx.x];
}

// ---------------------------------------------------------------------------
// Masked softmax over T + weighted sum s = alpha @ H.  grid = B
// ---------------------------------------------------------------------------
__global__ __launch_bounds__(256) void softmax_wsum_kernel(
    const float* __restrict__ scores, const float* __restrict__ H,
    const int* __restrict__ lens, float* __restrict__ out) {
  __shared__ float sAl[T_];
  __shared__ float red[256];
  const int b   = blockIdx.x;
  const int tid = threadIdx.x;
  const int len = lens[b];

  float mx = -1.0e30f;
  for (int t = tid; t < T_; t += 256) {
    float v = (t < len) ? scores[(size_t)b * T_ + t] : -1.0e30f;
    sAl[t] = v;
    mx = fmaxf(mx, v);
  }
  red[tid] = mx;
  __syncthreads();
  for (int s = 128; s > 0; s >>= 1) {
    if (tid < s) red[tid] = fmaxf(red[tid], red[tid + s]);
    __syncthreads();
  }
  mx = red[0];
  __syncthreads();

  float sum = 0.0f;
  for (int t = tid; t < T_; t += 256) {
    float e = __expf(sAl[t] - mx);
    sAl[t] = e;
    sum += e;
  }
  red[tid] = sum;
  __syncthreads();
  for (int s = 128; s > 0; s >>= 1) {
    if (tid < s) red[tid] += red[tid + s];
    __syncthreads();
  }
  const float inv = 1.0f / red[0];
  __syncthreads();
  for (int t = tid; t < T_; t += 256) sAl[t] *= inv;
  __syncthreads();

  for (int c = tid; c < DH; c += 256) {
    float s = 0.0f;
    for (int t = 0; t < T_; ++t) s += sAl[t] * H[((size_t)b * T_ + t) * DH + c];
    out[(size_t)b * DH + c] = s;
  }
}

// ---------------------------------------------------------------------------
// Host
// ---------------------------------------------------------------------------
extern "C" void kernel_launch(void* const* d_in, const int* in_sizes, int n_in,
                              void* d_out, int out_size, void* d_ws,
                              size_t ws_size, hipStream_t stream) {
  const float* words  = (const float*)d_in[0];
  const float* target = (const float*)d_in[1];
  const int*   lens   = (const int*)d_in[2];
  const float* Wih    = (const float*)d_in[3];
  const float* Whh    = (const float*)d_in[4];
  const float* bih    = (const float*)d_in[5];
  const float* bhh    = (const float*)d_in[6];
  const float* w1     = (const float*)d_in[7];
  const float* b1     = (const float*)d_in[8];
  const float* w      = (const float*)d_in[9];
  float* out = (float*)d_out;

  char*  ws  = (char*)d_ws;
  size_t off = 0;
  auto carve = [&](size_t bytes) -> void* {
    void* p = ws + off;
    off = (off + bytes + 255) & ~(size_t)255;
    return p;
  };
  __bf16* xw    = (__bf16*)carve((size_t)B_ * T_ * KX * sizeof(__bf16));
  __bf16* Wc    = (__bf16*)carve((size_t)NG * KTOT * sizeof(__bf16));
  float*  biasC = (float*)carve((size_t)NG * sizeof(float));
  __bf16* w1t   = (__bf16*)carve((size_t)DA * KATT * sizeof(__bf16));
  __bf16* aspb  = (__bf16*)carve((size_t)B_ * KX * sizeof(__bf16));
  __bf16* hb0   = (__bf16*)carve((size_t)B_ * DH * sizeof(__bf16));
  __bf16* hb1   = (__bf16*)carve((size_t)B_ * DH * sizeof(__bf16));
  int*    cnt   = (int*)carve(256);
  float*  Hbuf  = (float*)carve((size_t)B_ * T_ * DH * sizeof(float));
  float*  sbuf  = (float*)carve((size_t)B_ * T_ * sizeof(float));
  (void)ws_size; (void)in_sizes; (void)n_in; (void)out_size;

  {
    int n = B_ * T_ * KX;
    prep_words_kernel<<<(n + 255) / 256, 256, 0, stream>>>(words, xw, n);
  }
  {
    int n = NG * KTOT;
    prep_wcomb_kernel<<<(n + 255) / 256, 256, 0, stream>>>(Wih, Whh, Wc, n);
  }
  {
    int n = NG;
    prep_bias_kernel<<<(n + 255) / 256, 256, 0, stream>>>(bih, bhh, biasC, n);
  }
  {
    int n = DA * KATT;
    prep_w1t_kernel<<<(n + 255) / 256, 256, 0, stream>>>(w1, w1t, n);
  }
  {
    int n = B_ * KX;
    prep_aspect_kernel<<<(n + 255) / 256, 256, 0, stream>>>(target, aspb, n);
  }
  {
    int n = B_ * DH;
    prep_state_kernel<<<(n + 255) / 256, 256, 0, stream>>>(hb0, hb1, cnt, n);
  }

  // one persistent launch for all 512 serial steps (32 co-resident WGs)
  lstm_persistent_kernel<<<NBLK, 256, WLDS_BYTES, stream>>>(
      xw, Wc, biasC, lens, hb0, hb1, Hbuf, cnt);

  attn_score_kernel<<<(B_ * T_) / 16, 256, 0, stream>>>(Hbuf, aspb, w1t, b1, w,
                                                        sbuf);

  softmax_wsum_kernel<<<B_, 256, 0, stream>>>(sbuf, Hbuf, lens, out);
}
